// Dynamic_GCN_24232205484261
// MI455X (gfx1250) — compile-verified, matrix operand order checked by reading, not run on hardware
//
#include <hip/hip_runtime.h>
#include <math.h>

// ---------------------------------------------------------------------------
// Dynamic GCN (2 layers) for MI455X / gfx1250, bf16 WMMA everywhere.
// All GEMMs canonicalized to C = A * Y^T with A:[M,K], Y:[N,K] row-major bf16.
// LDS-staged block tile 128x128 (double-buffered, K-chunk 32); per-wave tile
// 32x64 (2x4 fragments, 8 WMMA per K-step). 256 threads = 8 waves (4x2).
// ---------------------------------------------------------------------------

typedef __attribute__((ext_vector_type(16))) __bf16       v16bf;
typedef __attribute__((ext_vector_type(8)))  float        v8f;
typedef __attribute__((ext_vector_type(4)))  unsigned int u32x4;

#define LAYERS 2
#define BATCH  16
#define NN     1024
#define DD     768

#define LDSROW     40                  // padded row stride (elements); 80B rows
#define TILE_ELEMS (128 * LDSROW)      // one 128x32 tile (padded)

__device__ __forceinline__ v8f wmma_bf16(v16bf a, v16bf b, v8f c) {
  return __builtin_amdgcn_wmma_f32_16x16x32_bf16(
      /*neg_a=*/false, a, /*neg_b=*/false, b,
      /*c_mod=*/(short)0, c, /*reuse_a=*/false, /*reuse_b=*/false);
}

// Fragment load from a [*, 32] row (stride LDSROW in LDS / arbitrary for tests).
// v16bf elements 0..7 = K = half*8 .. half*8+7 ; elements 8..15 = +16.
__device__ __forceinline__ v16bf load_frag(const __bf16* p, int half) {
  const u32x4* q = reinterpret_cast<const u32x4*>(p + half * 8);
  union { u32x4 u[2]; v16bf v; } x;
  x.u[0] = q[0];   // 16 bytes: K = half*8 .. +7
  x.u[1] = q[2];   // +32 bytes = +16 elems: K = half*8+16 .. +23
  return x.v;
}

struct Acc {
  v8f c[2][4];
};

// Block computes C[blkRow:+128, blkCol:+128] = A * Y^T staged through LDS.
// Wave w owns the 32x64 sub-tile at ((w>>1)*32, (w&1)*64).
__device__ __forceinline__ void gemm_tile_lds(const __bf16* __restrict__ A,
                                              const __bf16* __restrict__ Y,
                                              int lda, int ldb, int K,
                                              int blkRow, int blkCol,
                                              __bf16* smem, Acc& acc) {
  const int t = threadIdx.x;
  const int lane = t & 31;
  const int w = t >> 5;
  const int half = lane >> 4, lrow = lane & 15;
  const int crow = t >> 1;            // staging: row 0..127
  const int cseg = (t & 1) * 16;      // staging: element 0 or 16 within K-chunk

  const __bf16* gA = A + (size_t)(blkRow + crow) * lda + cseg;
  const __bf16* gB = Y + (size_t)(blkCol + crow) * ldb + cseg;
  const int sOff = crow * LDSROW + cseg;

  const int fAoff = ((w >> 1) * 32 + lrow) * LDSROW;
  const int fBoff = ((w & 1) * 64 + lrow) * LDSROW;

  u32x4 ra0, ra1, rb0, rb1;
  auto gload = [&](int k) {
    ra0 = *reinterpret_cast<const u32x4*>(gA + k);
    ra1 = *reinterpret_cast<const u32x4*>(gA + k + 8);
    rb0 = *reinterpret_cast<const u32x4*>(gB + k);
    rb1 = *reinterpret_cast<const u32x4*>(gB + k + 8);
    __builtin_prefetch(gA + k + 64, 0, 1);
    __builtin_prefetch(gB + k + 64, 0, 1);
  };
  auto sstore = [&](int buf) {
    __bf16* sA = smem + buf * (2 * TILE_ELEMS) + sOff;
    __bf16* sB = sA + TILE_ELEMS;
    *reinterpret_cast<u32x4*>(sA) = ra0;
    *reinterpret_cast<u32x4*>(sA + 8) = ra1;
    *reinterpret_cast<u32x4*>(sB) = rb0;
    *reinterpret_cast<u32x4*>(sB + 8) = rb1;
  };

  const int nIter = K / 32;
  gload(0);
  sstore(0);
  __syncthreads();
  for (int i = 0; i < nIter; ++i) {
    const int cur = i & 1;
    if (i + 1 < nIter) gload((i + 1) * 32);   // next stage -> regs (in flight)

    const __bf16* bA = smem + cur * (2 * TILE_ELEMS);
    const __bf16* bB = bA + TILE_ELEMS;
    v16bf fa[2], fb[4];
#pragma unroll
    for (int x = 0; x < 2; ++x)
      fa[x] = load_frag(bA + fAoff + x * 16 * LDSROW, half);
#pragma unroll
    for (int j = 0; j < 4; ++j)
      fb[j] = load_frag(bB + fBoff + j * 16 * LDSROW, half);
#pragma unroll
    for (int x = 0; x < 2; ++x)
#pragma unroll
      for (int j = 0; j < 4; ++j)
        acc.c[x][j] = wmma_bf16(fa[x], fb[j], acc.c[x][j]);

    if (i + 1 < nIter) sstore(cur ^ 1);
    __syncthreads();
  }
}

#define SMEM_DECL __shared__ __bf16 smem[2 * 2 * TILE_ELEMS]   // 40 KB

// --------------------------- elementwise kernels ---------------------------

__global__ __launch_bounds__(256) void k_cvt_bf16(const float* __restrict__ in,
                                                  __bf16* __restrict__ out,
                                                  size_t n) {
  size_t i = (size_t)blockIdx.x * blockDim.x + threadIdx.x;
  if (i < n) out[i] = (__bf16)in[i];
}

// AwT[l][e][d] = A_w[l][d][e], WwT likewise. idx is output-major (coalesced writes).
__global__ __launch_bounds__(256) void k_cvt_weights(const float* __restrict__ Aw,
                                                     const float* __restrict__ Ww,
                                                     __bf16* __restrict__ AwT,
                                                     __bf16* __restrict__ WwT,
                                                     size_t n) {
  size_t i = (size_t)blockIdx.x * blockDim.x + threadIdx.x;
  if (i >= n) return;
  size_t l = i / ((size_t)DD * DD);
  size_t r = i % ((size_t)DD * DD);
  size_t e = r / DD, d = r % DD;
  size_t src = l * DD * DD + d * DD + e;
  AwT[i] = (__bf16)Aw[src];
  WwT[i] = (__bf16)Ww[src];
}

__global__ __launch_bounds__(256) void k_dis(const float* __restrict__ deg,
                                             float* __restrict__ dis, int n) {
  int i = blockIdx.x * blockDim.x + threadIdx.x;
  if (i < n) {
    float d = deg[i];
    dis[i] = (d > 0.0f) ? rsqrtf(d) : 0.0f;
  }
}

// fsT[b][d][n] = bf16(dis[b,n] * feat[b,n,d])
__global__ __launch_bounds__(256) void k_fsT(const float* __restrict__ feat,
                                             const float* __restrict__ dis,
                                             __bf16* __restrict__ fsT) {
  size_t i = (size_t)blockIdx.x * blockDim.x + threadIdx.x;
  size_t b = i / ((size_t)NN * DD);
  size_t r = i % ((size_t)NN * DD);
  size_t n = r / DD, d = r % DD;
  fsT[b * (size_t)DD * NN + d * NN + n] = (__bf16)(dis[b * NN + n] * feat[i]);
}

// ------------------------------ GEMM kernels -------------------------------

// Plain: C(bf16) = A * Y^T   (used for G = F @ AwT)
__global__ __launch_bounds__(256) void k_gemm_plain(const __bf16* __restrict__ A,
                                                    const __bf16* __restrict__ Y,
                                                    __bf16* __restrict__ C,
                                                    int lda, int ldb, int ldc, int K) {
  SMEM_DECL;
  int lane = threadIdx.x & 31, w = threadIdx.x >> 5;
  int blkRow = blockIdx.x * 128, blkCol = blockIdx.y * 128;
  Acc acc = {};
  gemm_tile_lds(A, Y, lda, ldb, K, blkRow, blkCol, smem, acc);

  int half = lane >> 4, lrow = lane & 15;
  int row0 = blkRow + (w >> 1) * 32;
  int col0 = blkCol + (w & 1) * 64;
#pragma unroll
  for (int fx = 0; fx < 2; ++fx)
#pragma unroll
    for (int fy = 0; fy < 4; ++fy) {
      int col = col0 + fy * 16 + lrow;
#pragma unroll
      for (int i = 0; i < 8; ++i) {
        int row = row0 + fx * 16 + half * 8 + i;
        C[(size_t)row * ldc + col] = (__bf16)acc.c[fx][fy][i];
      }
    }
}

// Scores: S = (G_b @ F_b^T)/sqrt(D); adj = sigmoid(S)*mask; store adjT (bf16);
// accumulate deg (row sums) and per-batch stats {total, lastRow, lastCol, corner}.
__global__ __launch_bounds__(256) void k_scores(const __bf16* __restrict__ G,
                                                const __bf16* __restrict__ F,
                                                const float* __restrict__ pmask,
                                                __bf16* __restrict__ adjT,
                                                float* __restrict__ deg,
                                                float* __restrict__ statsL) {
  SMEM_DECL;
  const int b = blockIdx.z;
  const __bf16* A = G + (size_t)b * NN * DD;
  const __bf16* Y = F + (size_t)b * NN * DD;
  __bf16* adjTb = adjT + (size_t)b * NN * NN;

  int lane = threadIdx.x & 31, w = threadIdx.x >> 5;
  int blkRow = blockIdx.x * 128, blkCol = blockIdx.y * 128;
  Acc acc = {};
  gemm_tile_lds(A, Y, DD, DD, DD, blkRow, blkCol, smem, acc);

  const int half = lane >> 4, lrow = lane & 15;
  int row0 = blkRow + (w >> 1) * 32;
  int col0 = blkCol + (w & 1) * 64;
  const float inv_sqrt_d = 0.03608439182435161f;  // 1/sqrt(768)

  float pm_r[2][8], pm_c[4];
#pragma unroll
  for (int fx = 0; fx < 2; ++fx)
#pragma unroll
    for (int i = 0; i < 8; ++i)
      pm_r[fx][i] = pmask[(size_t)b * NN + row0 + fx * 16 + half * 8 + i];
#pragma unroll
  for (int fy = 0; fy < 4; ++fy)
    pm_c[fy] = pmask[(size_t)b * NN + col0 + fy * 16 + lrow];

  float rowsum[2][8];
#pragma unroll
  for (int fx = 0; fx < 2; ++fx)
#pragma unroll
    for (int i = 0; i < 8; ++i) rowsum[fx][i] = 0.0f;
  float tot = 0.0f, lastRow = 0.0f, lastCol = 0.0f, corner = 0.0f;

#pragma unroll
  for (int fx = 0; fx < 2; ++fx)
#pragma unroll
    for (int fy = 0; fy < 4; ++fy) {
      int col = col0 + fy * 16 + lrow;
#pragma unroll
      for (int i = 0; i < 8; ++i) {
        int row = row0 + fx * 16 + half * 8 + i;
        float s = acc.c[fx][fy][i] * inv_sqrt_d;
        float a = pm_r[fx][i] * pm_c[fy] / (1.0f + __expf(-s));
        adjTb[(size_t)col * NN + row] = (__bf16)a;  // transposed store
        rowsum[fx][i] += a;
        tot += a;
        if (row == NN - 1) lastRow += a;
        if (col == NN - 1) lastCol += a;
        if (row == NN - 1 && col == NN - 1) corner += a;
      }
    }

  // deg: reduce over the 16 lanes (columns) within each half.
  for (int m = 1; m < 16; m <<= 1)
#pragma unroll
    for (int fx = 0; fx < 2; ++fx)
#pragma unroll
      for (int i = 0; i < 8; ++i)
        rowsum[fx][i] += __shfl_xor(rowsum[fx][i], m, 32);
  if (lrow == 0)
#pragma unroll
    for (int fx = 0; fx < 2; ++fx)
#pragma unroll
      for (int i = 0; i < 8; ++i)
        atomicAdd(&deg[(size_t)b * NN + row0 + fx * 16 + half * 8 + i],
                  rowsum[fx][i]);

  // stats: full-wave reduction, lane 0 commits.
  for (int m = 1; m < 32; m <<= 1) {
    tot     += __shfl_xor(tot, m, 32);
    lastRow += __shfl_xor(lastRow, m, 32);
    lastCol += __shfl_xor(lastCol, m, 32);
    corner  += __shfl_xor(corner, m, 32);
  }
  if (lane == 0) {
    atomicAdd(&statsL[b * 4 + 0], tot);
    atomicAdd(&statsL[b * 4 + 1], lastRow);
    atomicAdd(&statsL[b * 4 + 2], lastCol);
    atomicAdd(&statsL[b * 4 + 3], corner);
  }
}

// Ax[b,m,d] = dis[b,m] * sum_n adjT[b,m,n] * fsT[b,d,n]   (out bf16)
__global__ __launch_bounds__(256) void k_gemm_ax(const __bf16* __restrict__ adjT,
                                                 const __bf16* __restrict__ fsT,
                                                 const float* __restrict__ dis,
                                                 __bf16* __restrict__ Ax) {
  SMEM_DECL;
  const int b = blockIdx.z;
  const __bf16* A = adjT + (size_t)b * NN * NN;
  const __bf16* Y = fsT + (size_t)b * DD * NN;

  int lane = threadIdx.x & 31, w = threadIdx.x >> 5;
  int blkRow = blockIdx.x * 128, blkCol = blockIdx.y * 128;
  Acc acc = {};
  gemm_tile_lds(A, Y, NN, NN, NN, blkRow, blkCol, smem, acc);

  int half = lane >> 4, lrow = lane & 15;
  int row0 = blkRow + (w >> 1) * 32;
  int col0 = blkCol + (w & 1) * 64;
#pragma unroll
  for (int fx = 0; fx < 2; ++fx)
#pragma unroll
    for (int fy = 0; fy < 4; ++fy) {
      int col = col0 + fy * 16 + lrow;
#pragma unroll
      for (int i = 0; i < 8; ++i) {
        int row = row0 + fx * 16 + half * 8 + i;
        float v = acc.c[fx][fy][i] * dis[(size_t)b * NN + row];
        Ax[((size_t)b * NN + row) * DD + col] = (__bf16)v;
      }
    }
}

// out(f32) = relu(Ax @ WwT + bias) + featIn    (M = BATCH*NN rows)
__global__ __launch_bounds__(256) void k_gemm_out(const __bf16* __restrict__ A,
                                                  const __bf16* __restrict__ Y,
                                                  const float* __restrict__ bias,
                                                  const float* __restrict__ featIn,
                                                  float* __restrict__ out,
                                                  int K) {
  SMEM_DECL;
  int lane = threadIdx.x & 31, w = threadIdx.x >> 5;
  int blkRow = blockIdx.x * 128, blkCol = blockIdx.y * 128;
  Acc acc = {};
  gemm_tile_lds(A, Y, K, K, K, blkRow, blkCol, smem, acc);

  int half = lane >> 4, lrow = lane & 15;
  int row0 = blkRow + (w >> 1) * 32;
  int col0 = blkCol + (w & 1) * 64;
#pragma unroll
  for (int fx = 0; fx < 2; ++fx)
#pragma unroll
    for (int fy = 0; fy < 4; ++fy) {
      int col = col0 + fy * 16 + lrow;
      float bcol = bias[col];
#pragma unroll
      for (int i = 0; i < 8; ++i) {
        int row = row0 + fx * 16 + half * 8 + i;
        float v = acc.c[fx][fy][i] + bcol;
        v = v > 0.0f ? v : 0.0f;
        out[(size_t)row * DD + col] = v + featIn[(size_t)row * DD + col];
      }
    }
}

// ------------------------------ stat finalize ------------------------------

__global__ void k_finalize(const float* __restrict__ pmask,
                           const float* __restrict__ stats,
                           float* __restrict__ outStats) {
  __shared__ float psum[BATCH];
  int t = threadIdx.x;
  if (t < BATCH) {
    float s = 0.0f;
    for (int n = 0; n < NN; ++n) s += pmask[(size_t)t * NN + n];
    psum[t] = s;
  }
  __syncthreads();
  if (t == 0) {
    float l0 = 0.0f;
    for (int l = 0; l < LAYERS; ++l) {
      float tsum = 0.0f;
      for (int b = 0; b < BATCH; ++b) tsum += stats[(l * BATCH + b) * 4 + 0];
      l0 += tsum / ((float)BATCH * NN * NN);
    }
    outStats[0] = l0 / (float)LAYERS;
    for (int b = 0; b < BATCH; ++b)
      for (int l = 0; l < LAYERS; ++l) {
        const float* s = &stats[(l * BATCH + b) * 4];
        float tot = s[0], lr = s[1], lc = s[2], cr = s[3];
        float ps = psum[b];
        outStats[1 + b * LAYERS + l] = (lr - cr) / ps;
        outStats[1 + BATCH * LAYERS + b * LAYERS + l] =
            (tot - lr - lc + cr) / (ps * ps);
      }
  }
}

// ------------------------------- launcher ----------------------------------

extern "C" void kernel_launch(void* const* d_in, const int* in_sizes, int n_in,
                              void* d_out, int out_size, void* d_ws, size_t ws_size,
                              hipStream_t stream) {
  const float* pmask   = (const float*)d_in[0];
  const float* feature = (const float*)d_in[1];
  const float* W_w     = (const float*)d_in[2];
  const float* W_b     = (const float*)d_in[3];
  const float* A_w     = (const float*)d_in[4];
  float* out = (float*)d_out;

  const size_t FSZ   = (size_t)BATCH * NN * DD;     // 12,582,912
  const size_t WSZ   = (size_t)LAYERS * DD * DD;    // per-weight total elems

  char* ws = (char*)d_ws;
  size_t off = 0;
  auto take = [&](size_t bytes) -> char* {
    char* p = ws + off;
    off += (bytes + 255) & ~(size_t)255;
    return p;
  };
  __bf16* buf1    = (__bf16*)take(sizeof(__bf16) * FSZ);               // fB / fsT
  __bf16* buf2    = (__bf16*)take(sizeof(__bf16) * FSZ);               // G  / Ax
  __bf16* adjT    = (__bf16*)take(sizeof(__bf16) * (size_t)BATCH * NN * NN);
  __bf16* AwT     = (__bf16*)take(sizeof(__bf16) * WSZ);
  __bf16* WwT     = (__bf16*)take(sizeof(__bf16) * WSZ);
  float*  featMid = (float*)take(sizeof(float) * FSZ);
  float*  deg     = (float*)take(sizeof(float) * BATCH * NN);
  float*  dis     = (float*)take(sizeof(float) * BATCH * NN);
  float*  stats   = (float*)take(sizeof(float) * LAYERS * BATCH * 4);

  // Weight transpose+convert (both layers, both matrices), stats zeroing.
  k_cvt_weights<<<(unsigned)((WSZ + 255) / 256), 256, 0, stream>>>(A_w, W_w, AwT, WwT, WSZ);
  hipMemsetAsync(stats, 0, sizeof(float) * LAYERS * BATCH * 4, stream);

  for (int l = 0; l < LAYERS; ++l) {
    const float* featIn  = (l == 0) ? feature : featMid;
    float*       featOut = (l == LAYERS - 1) ? out : featMid;
    const __bf16* AwTl = AwT + (size_t)l * DD * DD;
    const __bf16* WwTl = WwT + (size_t)l * DD * DD;
    const float*  Wbl  = W_b + (size_t)l * DD;
    float* statsL = stats + (size_t)l * BATCH * 4;

    // 1) fB = bf16(featIn)
    k_cvt_bf16<<<(unsigned)(FSZ / 256), 256, 0, stream>>>(featIn, buf1, FSZ);

    // 2) G = fB @ AwT   [B*N, D]
    k_gemm_plain<<<dim3(BATCH * NN / 128, DD / 128), 256, 0, stream>>>(
        buf1, AwTl, buf2, DD, DD, DD, DD);

    // 3) adjT / deg / stats
    hipMemsetAsync(deg, 0, sizeof(float) * BATCH * NN, stream);
    k_scores<<<dim3(NN / 128, NN / 128, BATCH), 256, 0, stream>>>(
        buf2, buf1, pmask, adjT, deg, statsL);

    // 4) dis = deg > 0 ? deg^-0.5 : 0
    k_dis<<<(BATCH * NN) / 256, 256, 0, stream>>>(deg, dis, BATCH * NN);

    // 5) fsT[b][d][n] = dis[b,n]*featIn[b,n,d]   (reuses buf1; fB is dead)
    k_fsT<<<(unsigned)(FSZ / 256), 256, 0, stream>>>(featIn, dis, buf1);

    // 6) Ax = dis[m] * (adjT @ fsT^T)            (reuses buf2; G is dead)
    k_gemm_ax<<<dim3(NN / 128, DD / 128, BATCH), 256, 0, stream>>>(
        adjT, buf1, dis, buf2);

    // 7) featOut = relu(Ax @ WwT + b) + featIn
    k_gemm_out<<<dim3(BATCH * NN / 128, DD / 128), 256, 0, stream>>>(
        buf2, WwTl, Wbl, featIn, featOut, DD);
  }

  // 8) scalar outputs: total_l0, c_sparsity[B,L], g_sparsity[B,L]
  k_finalize<<<1, 64, 0, stream>>>(pmask, stats, out + FSZ);
}
